// DyRepNode_70342974374371
// MI455X (gfx1250) — matrix operand order repaired and checked by reading, not compiled, not run on hardware
//
#include <hip/hip_runtime.h>
#include <hip/hip_bf16.h>
#include <math.h>

typedef __attribute__((ext_vector_type(16))) _Float16 v16h;
typedef __attribute__((ext_vector_type(8)))  float    v8f;

#define H      256
#define BATCH  512
#define DEG    16
#define NEG    5
#define NNODE  200000
#define TDMAX  86400.0f

// ---------------------------------------------------------------------------
// Kernel 1: parallel copy z0 -> z_final output region (float4, whole GPU)
// ---------------------------------------------------------------------------
__global__ __launch_bounds__(256) void copy_z0_kernel(const float4* __restrict__ src,
                                                      float4* __restrict__ dst, int n4) {
    int i = blockIdx.x * blockDim.x + threadIdx.x;
    if (i < n4) dst[i] = src[i];
}

// ---------------------------------------------------------------------------
// Kernel 2: convert 4 weight matrices (256x256 f32, row-major W[o][i]) into
// f16 WMMA B-fragments in d_ws.  Fragment layout (per 16x16x32 tile):
//   frag[g][kt][nt][lane][e], halfs;  n = nt*16 + (lane&15),
//   K = kt*32 + (lane>>4)*16 + e,  value = W[n][K]   (B = W^T)
// 65536 halfs (128 KB) per weight, 4 weights = 512 KB in d_ws.
// g order: 0=W_h, 1=W_rec_neigh, 2=W_rec_event, 3=W_e2n
// ---------------------------------------------------------------------------
__global__ __launch_bounds__(256) void convert_weights_kernel(
    const float* __restrict__ W0,
    const float* __restrict__ W1,
    const float* __restrict__ W2,
    const float* __restrict__ W3,
    _Float16* __restrict__ frag) {
    int tid = blockIdx.x * blockDim.x + threadIdx.x;   // 0 .. 4*65536-1
    if (tid >= 4 * 65536) return;
    int wsel = tid >> 16;
    int r    = tid & 65535;
    int e    = r & 15;
    int lane = (r >> 4) & 31;
    int rest = r >> 9;
    int nt   = rest & 15;
    int kt   = rest >> 4;                 // 0..7
    int n    = nt * 16 + (lane & 15);
    int K    = kt * 32 + ((lane >> 4) << 4) + e;
    const float* W = (wsel == 0) ? W0 : (wsel == 1) ? W1 : (wsel == 2) ? W2 : W3;
    frag[tid] = (_Float16)W[n * H + K];
}

// ---------------------------------------------------------------------------
// Kernel 3: persistent single-workgroup sequential scan (8 waves, wave32)
// Dynamic LDS: 256 KB = B-fragments of W_h (g0) and W_rec_neigh (g1).
// ---------------------------------------------------------------------------
__device__ __forceinline__ float sigmoidf(float x) { return 1.0f / (1.0f + __expf(-x)); }

__global__ __launch_bounds__(256) void dyrep_scan_kernel(
    const int*   __restrict__ u,          // (B,17)
    const float* __restrict__ time_delta, // (B,17,4)
    const float* __restrict__ time_bar,   // (B,N)
    const float* __restrict__ time_cur,   // (B,)
    const int*   __restrict__ u_neg,      // (B,5)
    const float* __restrict__ b_h,
    const float* __restrict__ b_e2n,
    const float* __restrict__ b_rec_event,
    const float* __restrict__ b_rec_neigh,
    const float* __restrict__ W_t,        // (H,4)
    const float* __restrict__ b_t,
    const float* __restrict__ W_omega,    // (1,H)
    const float* __restrict__ b_omega,    // (1,)
    const float* __restrict__ w_t_p, const float* __restrict__ alpha_p,
    const float* __restrict__ psi_p,
    const _Float16* __restrict__ wfrag,   // 4*65536 halfs (global, L2-resident)
    float* __restrict__ lam_pos,          // (B,)
    float* __restrict__ lam_neg,          // (B*5,)
    float* __restrict__ z) {              // (N,H), pre-initialized with z0

    const int tid  = threadIdx.x;      // 0..255
    const int lane = tid & 31;
    const int wave = tid >> 5;         // 0..7

    extern __shared__ _Float16 ldsB[];                 // 2*65536 halfs (256 KB)

    __shared__ __align__(32) _Float16 aZ[8][32][16];   // A-fragments of z_nb
    __shared__ __align__(32) _Float16 aE[8][32][16];   // A-fragments of [emb_u;0..]
    __shared__ float S[2][16][H];                      // GEMM results g0/g1
    __shared__ float vecE[H];                          // emb_u @ W_rec_event^T
    __shared__ float vecN[H];                          // emb_u @ W_e2n^T
    __shared__ float ebuf[6][H];                       // emb_u, emb_neg[0..4]
    __shared__ float prod[6][H];                       // hawkes staging
    __shared__ float tdn[17][4];
    __shared__ int   idx[17];
    __shared__ int   negidx[NEG];
    __shared__ float sc_tc;

    const float WT = w_t_p[0], AL = alpha_p[0], PS = psi_p[0], BO = b_omega[0];

    // ---- one-time: zero aE (rows M!=0 stay zero; M==0 rewritten each step) ----
    {
        _Float16* p = &aE[0][0][0];
        #pragma unroll
        for (int q = 0; q < 16; ++q) p[tid * 16 + q] = (_Float16)0.0f;
    }
    // ---- one-time: preload g0/g1 B-fragments into LDS (256 KB, 16B chunks) ----
    {
        const uint4* src = (const uint4*)wfrag;
        uint4* dst = (uint4*)ldsB;
        const int nchunk = (2 * 65536 * 2) / 16;       // 16384
        for (int c = tid; c < nchunk; c += 256) dst[c] = src[c];
    }

    // fragment addressing for K = tid (used in gather phase)
    const int gK   = tid;
    const int gkt  = gK >> 5;
    const int gko  = gK & 31;
    const int ge   = gko & 15;
    const int glo8 = gko >> 4;

    for (int step = 0; step < BATCH; ++step) {
        __syncthreads();   // prior-step scatters + one-time init visible

        // ----- phase A: indices, scalars, normalized time features -----
        if (tid < 17) idx[tid] = u[step * 17 + tid];
        if (tid < NEG) negidx[tid] = u_neg[step * NEG + tid];
        if (tid < 68) {
            int r = tid >> 2, c = tid & 3;
            float sd = (c == 0) ? 50.0f : (c == 1) ? 7.0f : 15.0f;
            tdn[r][c] = time_delta[step * 68 + tid] / sd;
        }
        if (tid == 0) sc_tc = time_cur[step];
        __syncthreads();

        // ----- phase B: gather state rows, convert to f16 A-fragments -----
        {
            #pragma unroll 4
            for (int j = 0; j < DEG; ++j) {
                float v = z[(size_t)idx[1 + j] * H + gK];
                aZ[gkt][j + (glo8 << 4)][ge] = (_Float16)v;
            }
            float vu = z[(size_t)idx[0] * H + gK];
            ebuf[0][gK] = vu;
            aE[gkt][(glo8 << 4)][ge] = (_Float16)vu;   // A row M=0
            #pragma unroll
            for (int i = 0; i < NEG; ++i)
                ebuf[1 + i][gK] = z[(size_t)negidx[i] * H + gK];
        }
        // prefetch next step's gather rows across the serial dependency
        if (step + 1 < BATCH) {
            if (tid < 17)
                __builtin_prefetch(&z[(size_t)u[(step + 1) * 17 + tid] * H], 0, 3);
            if (tid < NEG)
                __builtin_prefetch(&time_bar[(size_t)(step + 1) * NNODE +
                                             u_neg[(step + 1) * NEG + tid]], 0, 1);
        }
        __syncthreads();

        // ----- phase C: 64 WMMA tiles (4 GEMMs x 16 N-tiles), 8 per wave -----
        // Rolled outer loop: short live ranges, no register spilling.
        #pragma unroll 1
        for (int i = 0; i < 8; ++i) {
            int T  = wave + (i << 3);     // 0..63
            int g  = T >> 4;              // 0:W_h 1:W_rec_neigh 2:W_rec_event 3:W_e2n
            int nt = T & 15;
            v8f acc = {};
            if (g < 2) {
                // B from LDS (preloaded, latency-critical path)
                const _Float16* bf = ldsB + (g << 16) + ((nt * 32 + lane) << 4);
                #pragma unroll
                for (int kt = 0; kt < 8; ++kt) {
                    v16h a = *(const v16h*)&aZ[kt][lane][0];
                    v16h b = *(const v16h*)(bf + kt * 8192);
                    acc = __builtin_amdgcn_wmma_f32_16x16x32_f16(
                        false, a, false, b, (short)0, acc, false, false);
                }
            } else {
                // B streamed from L2-resident global fragments
                const _Float16* bf =
                    wfrag + ((size_t)g << 16) + ((size_t)(nt * 32 + lane) << 4);
                #pragma unroll
                for (int kt = 0; kt < 8; ++kt) {
                    v16h a = *(const v16h*)&aE[kt][lane][0];
                    v16h b = *(const v16h*)(bf + (size_t)kt * 8192);
                    acc = __builtin_amdgcn_wmma_f32_16x16x32_f16(
                        false, a, false, b, (short)0, acc, false, false);
                }
            }
            int nc    = nt * 16 + (lane & 15);
            int mhalf = lane >> 4;
            if (g < 2) {
                #pragma unroll
                for (int r = 0; r < 8; ++r) S[g][r + mhalf * 8][nc] = acc[r];
            } else if (mhalf == 0) {
                float* v = (g == 2) ? vecE : vecN;
                v[nc] = acc[0];           // only row M=0 is meaningful
            }
        }
        __syncthreads();

        // ----- phase D: elementwise update, scatter, hawkes staging -----
        {
            const int n = tid;
            float hs = 0.0f;
            #pragma unroll
            for (int j = 0; j < DEG; ++j) hs += S[0][j][n];
            hs = hs * (1.0f / 16.0f) + b_h[n];

            float wt0 = W_t[n * 4 + 0], wt1 = W_t[n * 4 + 1];
            float wt2 = W_t[n * 4 + 2], wt3 = W_t[n * 4 + 3];
            float bt  = b_t[n];

            float tf0 = tdn[0][0] * wt0 + tdn[0][1] * wt1 +
                        tdn[0][2] * wt2 + tdn[0][3] * wt3 + bt;
            float zev = sigmoidf(hs + vecE[n] + b_rec_event[n] + tf0);

            float ve = vecN[n] + b_e2n[n] + b_rec_neigh[n];
            #pragma unroll 4
            for (int j = 0; j < DEG; ++j) {
                float tf = tdn[1 + j][0] * wt0 + tdn[1 + j][1] * wt1 +
                           tdn[1 + j][2] * wt2 + tdn[1 + j][3] * wt3 + bt;
                float zn = sigmoidf(S[1][j][n] + ve + tf);
                z[(size_t)idx[1 + j] * H + n] = zn;   // neighbors first
            }
            z[(size_t)idx[0] * H + n] = zev;          // event row wins

            float wo = W_omega[n];
            #pragma unroll
            for (int i = 0; i < 6; ++i) prod[i][n] = ebuf[i][n] * wo;
        }
        __syncthreads();

        // ----- phase E: hawkes intensities (6 reductions) -----
        if (tid < 6) {
            float d = 0.0f;
            for (int c = 0; c < H; ++c) d += prod[tid][c];
            float ts;
            if (tid == 0) ts = sc_tc - tdn[0][0];
            else ts = sc_tc - time_bar[(size_t)step * NNODE + negidx[tid - 1]];
            float gg = d + BO + AL * __expf(-WT * (ts / TDMAX));
            float gp = fminf(fmaxf(gg / (PS + 1e-7f), -75.0f), 75.0f);
            float lam = PS * log1pf(__expf(gp));
            if (tid == 0) lam_pos[step] = lam;
            else          lam_neg[step * NEG + (tid - 1)] = lam / (float)NEG;
        }
    }
}

// ---------------------------------------------------------------------------
extern "C" void kernel_launch(void* const* d_in, const int* in_sizes, int n_in,
                              void* d_out, int out_size, void* d_ws, size_t ws_size,
                              hipStream_t stream) {
    const int*   u          = (const int*)  d_in[0];
    const float* time_delta = (const float*)d_in[1];
    const float* time_bar   = (const float*)d_in[2];
    const float* time_cur   = (const float*)d_in[3];
    // d_in[4] significance, d_in[5] magnitudo : unused by reference output
    const int*   u_neg      = (const int*)  d_in[6];
    const float* z0         = (const float*)d_in[7];
    const float* W_h        = (const float*)d_in[8];
    const float* b_h        = (const float*)d_in[9];
    const float* W_e2n      = (const float*)d_in[10];
    const float* b_e2n      = (const float*)d_in[11];
    const float* W_rec_ev   = (const float*)d_in[12];
    const float* b_rec_ev   = (const float*)d_in[13];
    const float* W_rec_nb   = (const float*)d_in[14];
    const float* b_rec_nb   = (const float*)d_in[15];
    const float* W_t        = (const float*)d_in[16];
    const float* b_t        = (const float*)d_in[17];
    const float* W_omega    = (const float*)d_in[18];
    const float* b_omega    = (const float*)d_in[19];
    const float* w_t        = (const float*)d_in[20];
    const float* alpha      = (const float*)d_in[21];
    const float* psi        = (const float*)d_in[22];

    float* out      = (float*)d_out;
    float* lam_pos  = out;                       // 512
    float* lam_neg  = out + BATCH;               // 2560
    float* z_final  = out + BATCH + BATCH * NEG; // 200000*256

    _Float16* wfrag = (_Float16*)d_ws;           // 512 KB of B-fragments

    // 1) z0 -> z_final (whole-GPU float4 copy; ~410 MB of HBM traffic)
    {
        int n4 = (NNODE * H) / 4;                // 12.8M float4
        int blocks = (n4 + 255) / 256;
        copy_z0_kernel<<<blocks, 256, 0, stream>>>(
            (const float4*)z0, (float4*)z_final, n4);
    }
    // 2) weights -> f16 WMMA B-fragments in workspace
    {
        int total = 4 * 65536;
        convert_weights_kernel<<<total / 256, 256, 0, stream>>>(
            W_h, W_rec_nb, W_rec_ev, W_e2n, wfrag);
    }
    // 3) persistent single-WGP sequential scan (256 KB dynamic LDS for weights)
    dyrep_scan_kernel<<<1, 256, 2 * 65536 * sizeof(_Float16), stream>>>(
        u, time_delta, time_bar, time_cur, u_neg,
        b_h, b_e2n, b_rec_ev, b_rec_nb,
        W_t, b_t, W_omega, b_omega, w_t, alpha, psi,
        wfrag, lam_pos, lam_neg, z_final);
}
